// VectorQuantizer_47914655154526
// MI455X (gfx1250) — matrix operand order repaired and checked by reading, not compiled, not run on hardware
//
#include <hip/hip_runtime.h>
#include <hip/hip_bf16.h>

// ---------------------------------------------------------------------------
// Vector quantizer for MI455X (gfx1250, wave32, WMMA).
//   inputs:  X [32768 x 256] f32,  W [8192 x 256] f32
//   outputs: quantized [32768 x 256] f32  ++  indices [32768] (as f32)
// argmin_k ( ||w_k||^2 - 2 x.w_k )  via v_wmma_f32_16x16x32_bf16 with
// 2-term bf16 splitting (hi*hi + hi*lo + lo*hi) for near-fp32 accuracy.
// Each wave owns TWO 16-row strips (32 rows) -> 48 WMMAs per 16KB LDS tile
// and two independent accumulator chains.  Per-iteration software pipeline:
//   issue global loads (tile c+1) -> WMMA compute (tile c) -> ds_store -> min.
// ---------------------------------------------------------------------------

typedef __attribute__((ext_vector_type(16))) __bf16 v16bf;
typedef __attribute__((ext_vector_type(8)))  float  v8f;

#define D_DIM   256
#define K_CB    8192
#define N_ROWS  32768
#define TILES   (K_CB / 16)        // 512 codeword tiles
#define KSTEPS  (D_DIM / 32)       // 8 wmma k-steps per tile

// ---------------------------------------------------------------------------
// Kernel 1: codeword squared norms.
// ---------------------------------------------------------------------------
__global__ __launch_bounds__(256) void vq_wnorm(const float* __restrict__ W,
                                                float* __restrict__ wnorm) {
  int k = blockIdx.x * 256 + threadIdx.x;
  if (k >= K_CB) return;
  const float4* row = (const float4*)(W + (size_t)k * D_DIM);
  float s = 0.f;
#pragma unroll 4
  for (int j = 0; j < D_DIM / 4; ++j) {
    float4 v = row[j];
    s = fmaf(v.x, v.x, s);
    s = fmaf(v.y, v.y, s);
    s = fmaf(v.z, v.z, s);
    s = fmaf(v.w, v.w, s);
  }
  wnorm[k] = s;
}

// ---------------------------------------------------------------------------
// Kernel 2: pack the codebook into WMMA B-fragment order, split bf16 hi/lo.
// Tile layout (16 KB per codeword tile c):
//   plane 0 (hi): 8 frags of 1 KB, frag s: lane l -> 32 bytes at l*32
//   plane 1 (lo): same, at +8 KB
// B-fragment element e (of v16bf), lane l (h=l/16, col j=l%16): e<8: K=8h+e ; e>=8: K=16+8h+(e-8).
// ---------------------------------------------------------------------------
__global__ __launch_bounds__(256) void vq_pack(const float* __restrict__ W,
                                               __bf16* __restrict__ pack) {
  int gid = blockIdx.x * 256 + threadIdx.x;     // [0, 2*K_CB*D_DIM)
  int c   = gid >> 13;                          // 8192 bf16 entries per tile
  int t   = gid & 8191;
  int p   = t >> 12;                            // 0 = hi, 1 = lo
  int rem = t & 4095;
  int s   = rem >> 9;                           // k-step
  int l   = (rem >> 4) & 31;                    // lane
  int e   = rem & 15;                           // element within v16bf
  int h   = l >> 4;
  int col = l & 15;                             // codeword within tile
  int kk  = (e < 8) ? 8 * h + e : 16 + 8 * h + (e - 8);  // B-matrix K index
  float x = W[(size_t)(c * 16 + col) * D_DIM + s * 32 + kk];
  __bf16 hi = (__bf16)x;
  pack[gid] = (p == 0) ? hi : (__bf16)(x - (float)hi);
}

// ---------------------------------------------------------------------------
// Kernel 3: WMMA argmin.  Block = 128 threads = 4 waves; each wave owns two
// 16-row strips (32 rows) -> block covers 128 rows, grid = 256 blocks.
// ---------------------------------------------------------------------------
__global__ __launch_bounds__(128) void vq_argmin(const float* __restrict__ X,
                                                 const float4* __restrict__ packTiles,
                                                 const float* __restrict__ wnorm,
                                                 int* __restrict__ idx_out) {
  __shared__ alignas(128) float4 lds[2][1024];  // 2 x 16 KB

  const int tid  = threadIdx.x;
  const int wave = tid >> 5;
  const int lane = tid & 31;
  const int r    = lane & 15;
  const int h    = lane >> 4;
  const int rowBase = blockIdx.x * 128 + wave * 32;   // strip0: +0, strip1: +16

  // ---- Load both strips' rows as split-bf16 A fragments ------------------
  // A layout (16x32 bf16): lane l -> row l%16; e<8: K=8h+e ; e>=8: K=16+8h+(e-8)
  v16bf a_hi[2][KSTEPS], a_lo[2][KSTEPS];
#pragma unroll
  for (int m = 0; m < 2; ++m) {
    const float* xrow = X + (size_t)(rowBase + 16 * m + r) * D_DIM;
#pragma unroll
    for (int s = 0; s < KSTEPS; ++s) {
      const float4* p0 = (const float4*)(xrow + s * 32 + 8 * h);        // K 8h..8h+7
      const float4* p1 = (const float4*)(xrow + s * 32 + 16 + 8 * h);   // K 16+8h..
      float4 q0 = p0[0], q1 = p0[1], q2 = p1[0], q3 = p1[1];
      float xs[16] = {q0.x, q0.y, q0.z, q0.w, q1.x, q1.y, q1.z, q1.w,
                      q2.x, q2.y, q2.z, q2.w, q3.x, q3.y, q3.z, q3.w};
#pragma unroll
      for (int e = 0; e < 16; ++e) {
        __bf16 hi = (__bf16)xs[e];
        a_hi[m][s][e] = hi;
        a_lo[m][s][e] = (__bf16)(xs[e] - (float)hi);
      }
    }
  }

  float minv[2][8];
  int   mini[2][8];
#pragma unroll
  for (int m = 0; m < 2; ++m)
#pragma unroll
    for (int i = 0; i < 8; ++i) { minv[m][i] = 3.4e38f; mini[m][i] = 0; }

  // ---- Preload tile 0 -----------------------------------------------------
#pragma unroll
  for (int j = 0; j < 8; ++j)
    lds[0][tid + 128 * j] = packTiles[tid + 128 * j];

  for (int c = 0; c < TILES; ++c) {
    __syncthreads();
    const int cur = c & 1;
    const bool haveNext = (c + 1 < TILES);

    // (1) Issue global loads for tile c+1 NOW; no waits until after compute.
    float4 t0, t1, t2, t3, t4, t5, t6, t7;
    if (haveNext) {
      const float4* src = packTiles + (size_t)(c + 1) * 1024;
      t0 = src[tid];       t1 = src[tid + 128]; t2 = src[tid + 256];
      t3 = src[tid + 384]; t4 = src[tid + 512]; t5 = src[tid + 640];
      t6 = src[tid + 768]; t7 = src[tid + 896];
    }
    // Prefetch tile c+2 into L2 (global_prefetch_b8); one cacheline/thread.
    if (c + 2 < TILES)
      __builtin_prefetch((const char*)(packTiles + (size_t)(c + 2) * 1024) + tid * 128,
                         0, 0);
    // Hoist this tile's codeword-norm load; latency hides behind WMMAs.
    const int k = c * 16 + r;
    const float wn = wnorm[k];

    // (2) Compute: two 16x16x256 score tiles, 48 bf16 WMMAs, 2 indep chains.
    v8f acc0 = {0.f, 0.f, 0.f, 0.f, 0.f, 0.f, 0.f, 0.f};
    v8f acc1 = {0.f, 0.f, 0.f, 0.f, 0.f, 0.f, 0.f, 0.f};
    const char* lp = (const char*)&lds[cur][0] + lane * 32;
#pragma unroll
    for (int s = 0; s < KSTEPS; ++s) {
      v16bf b_hi = *(const v16bf*)(lp + s * 1024);
      v16bf b_lo = *(const v16bf*)(lp + 8192 + s * 1024);
      acc0 = __builtin_amdgcn_wmma_f32_16x16x32_bf16(false, a_hi[0][s], false, b_hi,
                                                     (short)0, acc0, false, false);
      acc1 = __builtin_amdgcn_wmma_f32_16x16x32_bf16(false, a_hi[1][s], false, b_hi,
                                                     (short)0, acc1, false, false);
      acc0 = __builtin_amdgcn_wmma_f32_16x16x32_bf16(false, a_hi[0][s], false, b_lo,
                                                     (short)0, acc0, false, false);
      acc1 = __builtin_amdgcn_wmma_f32_16x16x32_bf16(false, a_hi[1][s], false, b_lo,
                                                     (short)0, acc1, false, false);
      acc0 = __builtin_amdgcn_wmma_f32_16x16x32_bf16(false, a_lo[0][s], false, b_hi,
                                                     (short)0, acc0, false, false);
      acc1 = __builtin_amdgcn_wmma_f32_16x16x32_bf16(false, a_lo[1][s], false, b_hi,
                                                     (short)0, acc1, false, false);
    }

    // (3) Drain staged tile c+1 into the other LDS buffer; the loadcnt waits
    //     are now covered by the 48 WMMAs above.
    if (haveNext) {
      lds[cur ^ 1][tid]       = t0;
      lds[cur ^ 1][tid + 128] = t1;
      lds[cur ^ 1][tid + 256] = t2;
      lds[cur ^ 1][tid + 384] = t3;
      lds[cur ^ 1][tid + 512] = t4;
      lds[cur ^ 1][tid + 640] = t5;
      lds[cur ^ 1][tid + 768] = t6;
      lds[cur ^ 1][tid + 896] = t7;
    }

    // (4) dist = wnorm[k] - 2*score ; running (min, argmin).
#pragma unroll
    for (int i = 0; i < 8; ++i) {
      float d0 = fmaf(-2.0f, acc0[i], wn);
      if (d0 < minv[0][i]) { minv[0][i] = d0; mini[0][i] = k; }
      float d1 = fmaf(-2.0f, acc1[i], wn);
      if (d1 < minv[1][i]) { minv[1][i] = d1; mini[1][i] = k; }
    }
  }

  // ---- reduce across the 16 lanes holding the same row -------------------
#pragma unroll
  for (int off = 1; off < 16; off <<= 1) {
#pragma unroll
    for (int m = 0; m < 2; ++m)
#pragma unroll
      for (int i = 0; i < 8; ++i) {
        float ov = __shfl_xor(minv[m][i], off, 32);
        int   oi = __shfl_xor(mini[m][i], off, 32);
        if (ov < minv[m][i] || (ov == minv[m][i] && oi < mini[m][i])) {
          minv[m][i] = ov; mini[m][i] = oi;
        }
      }
  }
  if (r == 0) {
#pragma unroll
    for (int m = 0; m < 2; ++m)
#pragma unroll
      for (int i = 0; i < 8; ++i)
        idx_out[rowBase + 16 * m + i + 8 * h] = mini[m][i];  // C row M = i + 8h
  }
}

// ---------------------------------------------------------------------------
// Kernel 4: gather quantized rows + emit indices.
// ---------------------------------------------------------------------------
__global__ __launch_bounds__(256) void vq_gather(const float* __restrict__ W,
                                                 const int* __restrict__ idx,
                                                 float* __restrict__ out_q,
                                                 float* __restrict__ out_idx) {
  int gid = blockIdx.x * 256 + threadIdx.x;
  if (gid < N_ROWS * D_DIM) {
    int row = gid >> 8;
    int d   = gid & 255;
    out_q[gid] = W[(size_t)idx[row] * D_DIM + d];
  }
  if (gid < N_ROWS) out_idx[gid] = (float)idx[gid];
}

// ---------------------------------------------------------------------------
extern "C" void kernel_launch(void* const* d_in, const int* in_sizes, int n_in,
                              void* d_out, int out_size, void* d_ws, size_t ws_size,
                              hipStream_t stream) {
  const float* X = (const float*)d_in[0];   // [32768 x 256]
  const float* W = (const float*)d_in[1];   // [8192 x 256]

  float* out_q   = (float*)d_out;                         // quantized
  float* out_idx = out_q + (size_t)N_ROWS * D_DIM;        // indices (as f32)

  // Workspace: wnorm (32 KB) | packed split-bf16 codebook (8 MB) | idx (128 KB)
  char*   ws    = (char*)d_ws;
  float*  wnorm = (float*)ws;
  __bf16* pack  = (__bf16*)(ws + 32 * 1024);
  int*    idxbf = (int*)(ws + 32 * 1024 + (size_t)8 * 1024 * 1024);

  vq_wnorm<<<K_CB / 256, 256, 0, stream>>>(W, wnorm);

  const int packElems = 2 * K_CB * D_DIM;   // hi + lo planes
  vq_pack<<<packElems / 256, 256, 0, stream>>>(W, pack);

  vq_argmin<<<N_ROWS / 128, 128, 0, stream>>>(X, (const float4*)pack, wnorm, idxbf);

  vq_gather<<<(N_ROWS * D_DIM) / 256, 256, 0, stream>>>(W, idxbf, out_q, out_idx);
}
